// UncertaintyGRU_84456236908595
// MI455X (gfx1250) — matrix operand order repaired and checked by reading, not compiled, not run on hardware
//
#include <hip/hip_runtime.h>

// ---------------------------------------------------------------------------
// MC-dropout 2-layer GRU ensemble for MI455X (gfx1250).
// 16 (model,pass) chains -> 16 persistent workgroups of 32 waves (wave32).
// Per-step GEMMs on v_wmma_f32_16x16x32_bf16; weights pre-swizzled into WMMA
// B-fragment layout (bf16, ~22 MB, L2-resident); f32 hidden state in
// registers; bf16 activations staged through LDS. Weight pointers are
// laundered per step (defeats LICM -> no spills) and returned as explicit
// global (AS1) pointers so fragment loads are global_load_b128, and the
// k-loops are double-buffered in source so loads overlap WMMA execution.
// ---------------------------------------------------------------------------

typedef __attribute__((ext_vector_type(16))) __bf16 v16bf;
typedef __attribute__((ext_vector_type(8)))  __bf16 v8bf;
typedef __attribute__((ext_vector_type(8)))  float  v8f;

typedef const __attribute__((address_space(1))) __bf16  gbf16;
typedef const __attribute__((address_space(1))) v16bf   gv16bf;

#define DEVI __device__ __forceinline__

constexpr int S = 256, B = 32, I = 256, H = 512, O = 256;
constexpr int Mm = 4, Pp = 4, NH3 = 3 * H;        // 1536
constexpr int KC_I = I / 32;                      // 8  k-chunks for K=I
constexpr int KC_H = H / 32;                      // 16 k-chunks for K=H
constexpr int NT3 = NH3 / 16;                     // 96 gate column tiles
constexpr int NT_O = O / 16;                      // 16 output column tiles

// swizzled fragment buffer sizes (bf16 elements)
constexpr size_t SZ_IH0 = (size_t)Mm * NT3 * KC_I * 512;   // w_ih0 (K=256)
constexpr size_t SZ_HH  = (size_t)Mm * NT3 * KC_H * 512;   // w_hh0/w_ih1/w_hh1
constexpr size_t SZ_LIN = (size_t)Mm * NT_O * KC_H * 512;  // lin_w

DEVI v8f wmma_bf16(v16bf a, v16bf b, v8f c) {
    return __builtin_amdgcn_wmma_f32_16x16x32_bf16(
        /*neg_a=*/false, a, /*neg_b=*/false, b,
        /*c_mod=*/(short)0, c, /*reuse_a=*/false, /*reuse_b=*/false);
}

// Opaque copy of a uniform pointer (blocks LICM of the fragment loads), cast
// back to an explicit global-address-space pointer so loads lower to
// global_load_b128 (LOADcnt only) instead of flat_load_b128.
DEVI gbf16* launder(const __bf16* p) {
    asm volatile("" : "+s"(p));
    return (gbf16*)p;
}

// B-fragment: pre-swizzled so each lane reads 16 contiguous bf16 (32 B).
DEVI v16bf bfrag(gbf16* w, int tile, int kc, int nk, int lane) {
    return *(gv16bf*)(w + (((size_t)tile * nk + kc) * 32 + lane) * 16);
}

// A-fragment from LDS (row-major bf16 activations [32][stride]).
// lane l -> row m=(l&15)+16*mt; elems 0..7 = K[(l>>4)*8 ..],
// elems 8..15 = K[16+(l>>4)*8 ..] within the 32-wide k-chunk.
DEVI v16bf lds_afrag(const __bf16* buf, int stride, int mt, int kc, int lane) {
    const __bf16* p = buf + (size_t)((lane & 15) + mt * 16) * stride
                          + kc * 32 + ((lane >> 4) * 8);
    v8bf lo = *(const v8bf*)(p);
    v8bf hi = *(const v8bf*)(p + 16);
    return __builtin_shufflevector(lo, hi, 0,1,2,3,4,5,6,7,8,9,10,11,12,13,14,15);
}

DEVI float sigmoidf_(float v) { return 1.0f / (1.0f + __expf(-v)); }

// Three-gate GEMM slab, double-buffered over k-chunks: loads for chunk kc+1
// are issued before the WMMAs of chunk kc so L2 latency overlaps the matrix
// pipe instead of serializing on s_wait loadcnt 0.
template<int NK>
DEVI void gemm3(const __bf16* abuf, int astride, int mt, int lane,
                gbf16* W, int t_r, int t_z, int t_n,
                v8f& accr, v8f& accz, v8f& accn) {
    v16bf a  = lds_afrag(abuf, astride, mt, 0, lane);
    v16bf br = bfrag(W, t_r, 0, NK, lane);
    v16bf bz = bfrag(W, t_z, 0, NK, lane);
    v16bf bn = bfrag(W, t_n, 0, NK, lane);
    #pragma unroll
    for (int kc = 0; kc < NK; ++kc) {
        v16bf a2 = a, br2 = br, bz2 = bz, bn2 = bn;
        if (kc + 1 < NK) {
            a2  = lds_afrag(abuf, astride, mt, kc + 1, lane);
            br2 = bfrag(W, t_r, kc + 1, NK, lane);
            bz2 = bfrag(W, t_z, kc + 1, NK, lane);
            bn2 = bfrag(W, t_n, kc + 1, NK, lane);
        }
        accr = wmma_bf16(a, br, accr);
        accz = wmma_bf16(a, bz, accz);
        accn = wmma_bf16(a, bn, accn);
        a = a2; br = br2; bz = bz2; bn = bn2;
    }
}

// Single-tile GEMM (output head), same double-buffering.
template<int NK>
DEVI void gemm1(const __bf16* abuf, int astride, int mt, int lane,
                gbf16* W, int tile, v8f& acc) {
    v16bf a = lds_afrag(abuf, astride, mt, 0, lane);
    v16bf b = bfrag(W, tile, 0, NK, lane);
    #pragma unroll
    for (int kc = 0; kc < NK; ++kc) {
        v16bf a2 = a, b2 = b;
        if (kc + 1 < NK) {
            a2 = lds_afrag(abuf, astride, mt, kc + 1, lane);
            b2 = bfrag(W, tile, kc + 1, NK, lane);
        }
        acc = wmma_bf16(a, b, acc);
        a = a2; b = b2;
    }
}

// ---------------------------------------------------------------------------
// Weight swizzle: [rows,K] f32 row-major -> bf16 WMMA B-fragments.
// ---------------------------------------------------------------------------
__global__ void swizzle_w(const float* __restrict__ w, __bf16* __restrict__ dst,
                          int rows, int K, long total) {
    const int nk = K / 32;
    const long perModel = (long)(rows / 16) * nk * 512;
    for (long e = blockIdx.x * (long)blockDim.x + threadIdx.x; e < total;
         e += (long)gridDim.x * blockDim.x) {
        int  mdl   = (int)(e / perModel);
        long r1    = e % perModel;
        int  ln16  = (int)(r1 % 512);
        int  frag  = (int)(r1 / 512);
        int  tile  = frag / nk, kc = frag % nk;
        int  lane  = ln16 >> 4, j = ln16 & 15;
        int  row   = tile * 16 + (lane & 15);
        int  k     = kc * 32 + ((lane >> 4) * 8) + (j & 7) + ((j >> 3) * 16);
        dst[e] = (__bf16)w[(size_t)mdl * rows * K + (size_t)row * K + k];
    }
}

// ---------------------------------------------------------------------------
// Persistent per-chain GRU kernel: block = one (model,pass), 32 waves.
// ---------------------------------------------------------------------------
__global__ __launch_bounds__(1024, 1)
void gru_persistent(const float* __restrict__ x,       // [S,B,I]
                    const float* __restrict__ h_init,  // [M,P,L,B,H]
                    const float* __restrict__ b_ih0, const float* __restrict__ b_hh0,
                    const float* __restrict__ b_ih1, const float* __restrict__ b_hh1,
                    const float* __restrict__ lin_b,   // [M,O]
                    const float* __restrict__ mask,    // [M,P,S,B,H]
                    const __bf16* __restrict__ wfih0, const __bf16* __restrict__ wfhh0,
                    const __bf16* __restrict__ wfih1, const __bf16* __restrict__ wfhh1,
                    const __bf16* __restrict__ wflin,
                    float* __restrict__ out)           // [mean|var|preds|hidden]
{
    __shared__ __align__(32) __bf16 xb[32][256];   // x_t (bf16)        16 KB
    __shared__ __align__(32) __bf16 hb0[32][512];  // h layer0 (bf16)   32 KB
    __shared__ __align__(32) __bf16 hb1[32][512];  // h layer1 (bf16)   32 KB
    __shared__ __align__(32) __bf16 yb[32][512];   // y0*mask / act(y1) 32 KB

    const int tid  = threadIdx.x;
    const int lane = tid & 31;
    const int wv   = tid >> 5;                 // 0..31
    const int blk  = blockIdx.x;               // 0..15
    const int mdl  = blk >> 2, ps = blk & 3;
    const int mp   = mdl * Pp + ps;

    const __bf16* WI0 = wfih0 + (size_t)mdl * NT3 * KC_I * 512;
    const __bf16* WH0 = wfhh0 + (size_t)mdl * NT3 * KC_H * 512;
    const __bf16* WI1 = wfih1 + (size_t)mdl * NT3 * KC_H * 512;
    const __bf16* WH1 = wfhh1 + (size_t)mdl * NT3 * KC_H * 512;
    const __bf16* WL  = wflin + (size_t)mdl * NT_O * KC_H * 512;

    const int colH  = wv * 16 + (lane & 15);   // this wave's H column
    const int rbase = (lane >> 4) * 8;         // row base inside a 16-row tile

    // gate biases for this column (uniform over the 8 accumulator rows)
    const float bi0r = b_ih0[mdl*NH3 + 0*H + colH], bh0r = b_hh0[mdl*NH3 + 0*H + colH];
    const float bi0z = b_ih0[mdl*NH3 + 1*H + colH], bh0z = b_hh0[mdl*NH3 + 1*H + colH];
    const float bi0n = b_ih0[mdl*NH3 + 2*H + colH], bh0n = b_hh0[mdl*NH3 + 2*H + colH];
    const float bi1r = b_ih1[mdl*NH3 + 0*H + colH], bh1r = b_hh1[mdl*NH3 + 0*H + colH];
    const float bi1z = b_ih1[mdl*NH3 + 1*H + colH], bh1z = b_hh1[mdl*NH3 + 1*H + colH];
    const float bi1n = b_ih1[mdl*NH3 + 2*H + colH], bh1n = b_hh1[mdl*NH3 + 2*H + colH];

    // persistent f32 hidden state (D-fragment layout), 2 M-tiles per layer
    v8f h0s[2], h1s[2];
    {
        const size_t base = (size_t)mp * 2 /*L*/ * B * H;
        for (int mt = 0; mt < 2; ++mt)
            for (int vi = 0; vi < 8; ++vi) {
                int brow = mt * 16 + rbase + vi;
                h0s[mt][vi] = h_init[base + (size_t)brow * H + colH];
                h1s[mt][vi] = h_init[base + (size_t)B * H + (size_t)brow * H + colH];
            }
        for (int mt = 0; mt < 2; ++mt)
            for (int vi = 0; vi < 8; ++vi) {
                int brow = mt * 16 + rbase + vi;
                hb0[brow][colH] = (__bf16)h0s[mt][vi];
                hb1[brow][colH] = (__bf16)h1s[mt][vi];
            }
    }
    __syncthreads();

    const size_t maskMP = (size_t)mp * S * B * H;
    float* preds = out + 2 * (size_t)S * B * O + (size_t)mp * S * B * O;

    #pragma unroll 1
    for (int t = 0; t < S; ++t) {
        // Re-launder weight pointers every step (blocks LICM; loads stay
        // in-loop as global_load_b128 served from L2).
        gbf16* WI0t = launder(WI0);
        gbf16* WH0t = launder(WH0);
        gbf16* WI1t = launder(WI1);
        gbf16* WH1t = launder(WH1);
        gbf16* WLt  = launder(WL);

        // stage x_t -> bf16 LDS
        const float* xt = x + (size_t)t * B * I;
        for (int i = tid; i < B * I; i += 1024)
            (&xb[0][0])[i] = (__bf16)xt[i];
        __syncthreads();

        // -------- layer 0: gates = x_t@Wih0^T + h0@Whh0^T --------
        for (int mt = 0; mt < 2; ++mt) {
            v8f ar = {}, az = {}, ain = {}, ahn = {};
            gemm3<KC_I>(&xb[0][0], 256, mt, lane, WI0t, wv, 32 + wv, 64 + wv,
                        ar, az, ain);
            gemm3<KC_H>(&hb0[0][0], 512, mt, lane, WH0t, wv, 32 + wv, 64 + wv,
                        ar, az, ahn);
            #pragma unroll
            for (int vi = 0; vi < 8; ++vi) {
                float r = sigmoidf_(ar[vi] + bi0r + bh0r);
                float z = sigmoidf_(az[vi] + bi0z + bh0z);
                float n = tanhf(ain[vi] + bi0n + r * (ahn[vi] + bh0n));
                h0s[mt][vi] = (1.0f - z) * n + z * h0s[mt][vi];
            }
        }
        __syncthreads();   // all hb0 reads done before overwrite
        {
            const float* mk = mask + maskMP + (size_t)t * B * H;
            for (int mt = 0; mt < 2; ++mt)
                for (int vi = 0; vi < 8; ++vi) {
                    int brow = mt * 16 + rbase + vi;
                    float h = h0s[mt][vi];
                    hb0[brow][colH] = (__bf16)h;
                    yb[brow][colH]  = (__bf16)(h * mk[(size_t)brow * H + colH]);
                }
        }
        __syncthreads();

        // -------- layer 1: gates = (y0*mask)@Wih1^T + h1@Whh1^T --------
        for (int mt = 0; mt < 2; ++mt) {
            v8f ar = {}, az = {}, ain = {}, ahn = {};
            gemm3<KC_H>(&yb[0][0], 512, mt, lane, WI1t, wv, 32 + wv, 64 + wv,
                        ar, az, ain);
            gemm3<KC_H>(&hb1[0][0], 512, mt, lane, WH1t, wv, 32 + wv, 64 + wv,
                        ar, az, ahn);
            #pragma unroll
            for (int vi = 0; vi < 8; ++vi) {
                float r = sigmoidf_(ar[vi] + bi1r + bh1r);
                float z = sigmoidf_(az[vi] + bi1z + bh1z);
                float n = tanhf(ain[vi] + bi1n + r * (ahn[vi] + bh1n));
                h1s[mt][vi] = (1.0f - z) * n + z * h1s[mt][vi];
            }
        }
        __syncthreads();   // yb/hb1 reads done
        for (int mt = 0; mt < 2; ++mt)
            for (int vi = 0; vi < 8; ++vi) {
                int brow = mt * 16 + rbase + vi;
                float h = h1s[mt][vi];
                hb1[brow][colH] = (__bf16)h;
                yb[brow][colH]  = (__bf16)(h >= 0.0f ? h : 0.01f * h);  // LeakyReLU
            }
        __syncthreads();

        // -------- output head: preds[t] = act @ lin_w^T + lin_b --------
        {
            int mtO = wv >> 4, ntO = wv & 15;      // 32 waves = 2x16 out tiles
            v8f acc = {};
            gemm1<KC_H>(&yb[0][0], 512, mtO, lane, WLt, ntO, acc);
            int ocol = ntO * 16 + (lane & 15);
            float lb = lin_b[mdl * O + ocol];
            float* pt = preds + (size_t)t * B * O;
            #pragma unroll
            for (int vi = 0; vi < 8; ++vi) {
                int brow = mtO * 16 + rbase + vi;
                pt[(size_t)brow * O + ocol] = acc[vi] + lb;
            }
        }
    }

    // final hidden state -> out[... hidden region], layout [M,P,L,B,H]
    {
        float* hout = out + 2 * (size_t)S * B * O + (size_t)Mm * Pp * S * B * O
                    + (size_t)mp * 2 * B * H;
        for (int mt = 0; mt < 2; ++mt)
            for (int vi = 0; vi < 8; ++vi) {
                int brow = mt * 16 + rbase + vi;
                hout[(size_t)brow * H + colH]                 = h0s[mt][vi];
                hout[(size_t)B * H + (size_t)brow * H + colH] = h1s[mt][vi];
            }
    }
}

// ---------------------------------------------------------------------------
// mean / unbiased variance over the 16 (model,pass) predictions
// ---------------------------------------------------------------------------
__global__ void mean_var(const float* __restrict__ preds, float* __restrict__ mean,
                         float* __restrict__ var, long n) {
    long e = blockIdx.x * (long)blockDim.x + threadIdx.x;
    if (e >= n) return;
    float s = 0.0f, sq = 0.0f;
    #pragma unroll
    for (int mpi = 0; mpi < Mm * Pp; ++mpi) {
        float v = preds[(size_t)mpi * n + e];
        s += v; sq += v * v;
    }
    float mu = s * (1.0f / 16.0f);
    mean[e] = mu;
    var[e]  = (sq - 16.0f * mu * mu) * (1.0f / 15.0f);
}

extern "C" void kernel_launch(void* const* d_in, const int* in_sizes, int n_in,
                              void* d_out, int out_size, void* d_ws, size_t ws_size,
                              hipStream_t stream) {
    const float* x     = (const float*)d_in[0];
    const float* hid   = (const float*)d_in[1];
    const float* w_ih0 = (const float*)d_in[2];
    const float* w_hh0 = (const float*)d_in[3];
    const float* b_ih0 = (const float*)d_in[4];
    const float* b_hh0 = (const float*)d_in[5];
    const float* w_ih1 = (const float*)d_in[6];
    const float* w_hh1 = (const float*)d_in[7];
    const float* b_ih1 = (const float*)d_in[8];
    const float* b_hh1 = (const float*)d_in[9];
    const float* lin_w = (const float*)d_in[10];
    const float* lin_b = (const float*)d_in[11];
    const float* mask  = (const float*)d_in[12];
    float* out = (float*)d_out;

    __bf16* wf = (__bf16*)d_ws;
    __bf16* f_ih0 = wf;                 size_t o = SZ_IH0;
    __bf16* f_hh0 = wf + o;             o += SZ_HH;
    __bf16* f_ih1 = wf + o;             o += SZ_HH;
    __bf16* f_hh1 = wf + o;             o += SZ_HH;
    __bf16* f_lin = wf + o;             // total ~23 MB of workspace

    auto sw = [&](const float* w, __bf16* dst, int rows, int K) {
        long total = (long)Mm * (rows / 16) * (K / 32) * 512;
        int blocks = (int)((total + 255) / 256);
        swizzle_w<<<blocks, 256, 0, stream>>>(w, dst, rows, K, total);
    };
    sw(w_ih0, f_ih0, NH3, I);
    sw(w_hh0, f_hh0, NH3, H);
    sw(w_ih1, f_ih1, NH3, H);
    sw(w_hh1, f_hh1, NH3, H);
    sw(lin_w, f_lin, O,   H);

    gru_persistent<<<dim3(Mm * Pp), dim3(1024), 0, stream>>>(
        x, hid, b_ih0, b_hh0, b_ih1, b_hh1, lin_b, mask,
        f_ih0, f_hh0, f_ih1, f_hh1, f_lin, out);

    long n = (long)S * B * O;
    mean_var<<<dim3((unsigned)((n + 255) / 256)), dim3(256), 0, stream>>>(
        out + 2 * n, out, out + n, n);
}
